// CIND_Block_50843822850593
// MI455X (gfx1250) — matrix operand order repaired and checked by reading, not compiled, not run on hardware
//
#include <hip/hip_runtime.h>
#include <hip/hip_bf16.h>

typedef __attribute__((ext_vector_type(16))) _Float16 v16h;
typedef __attribute__((ext_vector_type(8)))  float    v8f;

// ---------------- fragment layout maps (CDNA5 WMMA 16-bit, wave32) ----------
// A-matrix 16x32 (MxK): lanes 0-15 row M=lane, halves h=0..7 -> K=h, h=8..15 -> K=h+8;
// lanes 16-31 same rows, +8 on K.
__device__ __forceinline__ int kmap_a(int lane, int h) {
    return h + ((h & 8) ? 8 : 0) + ((lane & 16) ? 8 : 0);
}
// B-matrix 32x16 (KxN): lanes 0-15 col N=lane, h=0..7 -> K=h, h=8..15 -> K=h+8;
// lanes 16-31 same cols, +8 on K.
__device__ __forceinline__ int kmap_b(int lane, int h) {
    return h + ((h & 8) ? 8 : 0) + ((lane & 16) ? 8 : 0);
}

// ---------------- problem constants ----------------------------------------
#define Cc   256
#define Bb   64
#define K1   6656   // 26*256
#define KT1  208
#define N1   3136   // 64*49
#define NT1  196
#define K23  2304   // 9*256
#define KT23 72
#define N2   1600   // 64*25
#define NT2  100
#define N3   576    // 64*9
#define NT3  36

// ---------------- packing kernels -------------------------------------------
// A1[m][k], k = c*26 + t. t<25 -> -w1[m,c,t]; t==25 -> S1[m,c] = sum_t w1[m,c,t].
__global__ void pack_a1(const float* __restrict__ w1, _Float16* __restrict__ Ap) {
    int gid = blockIdx.x * 256 + threadIdx.x;
    if (gid >= 16 * KT1 * 32 * 16) return;
    int h = gid & 15, lane = (gid >> 4) & 31, rest = gid >> 9;
    int kt = rest % KT1, mt = rest / KT1;
    int m = mt * 16 + (lane & 15);
    int k = kt * 32 + kmap_a(lane, h);
    int c = k / 26, t = k - c * 26;
    float v;
    const float* wp = w1 + (m * Cc + c) * 25;
    if (t < 25) {
        v = -wp[t];
    } else {
        v = 0.f;
        for (int u = 0; u < 25; ++u) v += wp[u];
    }
    Ap[gid] = (_Float16)v;
}

// B1[k][n], n = b*49 + i*7 + j. t<25: x_r_pad[b,c,i+a-2,j+b'-2]; t==25: x_l[b,c,i,j].
__global__ void pack_b1(const float* __restrict__ x_r, const float* __restrict__ x_l,
                        _Float16* __restrict__ Bp) {
    int gid = blockIdx.x * 256 + threadIdx.x;
    if (gid >= NT1 * KT1 * 32 * 16) return;
    int h = gid & 15, lane = (gid >> 4) & 31, rest = gid >> 9;
    int kt = rest % KT1, nt = rest / KT1;
    int n = nt * 16 + (lane & 15);
    int k = kt * 32 + kmap_b(lane, h);
    int c = k / 26, t = k - c * 26;
    int bidx = n / 49, p = n % 49, i = p / 7, j = p % 7;
    float v;
    if (t < 25) {
        int a = t / 5, bo = t % 5;
        int ri = i + a - 2, rj = j + bo - 2;
        v = (ri >= 0 && ri < 7 && rj >= 0 && rj < 7)
              ? x_r[(bidx * Cc + c) * 49 + ri * 7 + rj] : 0.f;
    } else {
        v = x_l[(bidx * Cc + c) * 49 + i * 7 + j];
    }
    Bp[gid] = (_Float16)v;
}

// A for the 3x3 convs: A[m][k], k = c*9 + t, value = w[m,c,t].
__global__ void pack_a_3x3(const float* __restrict__ w, _Float16* __restrict__ Ap) {
    int gid = blockIdx.x * 256 + threadIdx.x;
    if (gid >= 16 * KT23 * 32 * 16) return;
    int h = gid & 15, lane = (gid >> 4) & 31, rest = gid >> 9;
    int kt = rest % KT23, mt = rest / KT23;
    int m = mt * 16 + (lane & 15);
    int k = kt * 32 + kmap_a(lane, h);
    int c = k / 9, t = k - c * 9;
    Ap[gid] = (_Float16)w[(m * Cc + c) * 9 + t];
}

// B2: im2col of y1 (f32, layout [C][64*49]) for 3x3 VALID conv 7x7 -> 5x5.
__global__ void pack_b2(const float* __restrict__ y1, _Float16* __restrict__ Bp) {
    int gid = blockIdx.x * 256 + threadIdx.x;
    if (gid >= NT2 * KT23 * 32 * 16) return;
    int h = gid & 15, lane = (gid >> 4) & 31, rest = gid >> 9;
    int kt = rest % KT23, nt = rest / KT23;
    int n = nt * 16 + (lane & 15);
    int k = kt * 32 + kmap_b(lane, h);
    int c = k / 9, t = k - c * 9, a = t / 3, bo = t % 3;
    int bidx = n / 25, p = n % 25, i = p / 5, j = p % 5;
    Bp[gid] = (_Float16)y1[(size_t)c * N1 + bidx * 49 + (i + a) * 7 + (j + bo)];
}

// B3: im2col of y2 (f32, layout [C][64*25]) for 3x3 VALID conv 5x5 -> 3x3.
__global__ void pack_b3(const float* __restrict__ y2, _Float16* __restrict__ Bp) {
    int gid = blockIdx.x * 256 + threadIdx.x;
    if (gid >= NT3 * KT23 * 32 * 16) return;
    int h = gid & 15, lane = (gid >> 4) & 31, rest = gid >> 9;
    int kt = rest % KT23, nt = rest / KT23;
    int n = nt * 16 + (lane & 15);
    int k = kt * 32 + kmap_b(lane, h);
    int c = k / 9, t = k - c * 9, a = t / 3, bo = t % 3;
    int bidx = n / 9, p = n % 9, i = p / 3, j = p % 3;
    Bp[gid] = (_Float16)y2[(size_t)c * N2 + bidx * 25 + (i + a) * 5 + (j + bo)];
}

// ---------------- WMMA GEMM + bias + ReLU -----------------------------------
// Block = 4 waves (128 threads). All waves share one N-tile; wave w computes
// M-tile (blockIdx.x*4 + w). The shared 32x16 f16 B fragment (1 KB) for each
// k-step is staged into LDS with async global->LDS copies (ASYNCcnt),
// double-buffered; A fragments are loaded directly from global with prefetch.
__global__ void __launch_bounds__(128)
wmma_gemm_relu(const _Float16* __restrict__ Ap, const _Float16* __restrict__ Bp,
               const float* __restrict__ bias, float* __restrict__ Y,
               int KT, int Ntot) {
    __shared__ _Float16 shB[2][512];           // 2 x 1 KB double buffer
    const int nt   = blockIdx.y;
    const int tid  = threadIdx.x;              // 0..127
    const int wave = tid >> 5;                 // 0..3
    const int lane = tid & 31;
    const int mt   = blockIdx.x * 4 + wave;

    const v16h* a = (const v16h*)Ap + (size_t)mt * KT * 32 + lane;
    const char* bsrc = (const char*)(Bp + (size_t)nt * KT * 512);  // 1 KB per k-step

    // LDS byte offsets for this thread's 8-byte slice of each buffer
    const unsigned lds0 = (unsigned)(uintptr_t)(&shB[0][0]) + tid * 8;
    const unsigned lds1 = (unsigned)(uintptr_t)(&shB[1][0]) + tid * 8;

    // prologue: async-copy B tile kt=0 into buffer 0 (128 threads x 8 B = 1 KB)
    {
        const char* g = bsrc + tid * 8;
        asm volatile("global_load_async_to_lds_b64 %0, %1, off"
                     :: "v"(lds0), "v"(g) : "memory");
    }

    v8f acc = {};
    for (int kt = 0; kt < KT; ++kt) {
        if (kt + 1 < KT) {
            // async-copy next B tile into the other buffer
            const unsigned ldsn = ((kt + 1) & 1) ? lds1 : lds0;
            const char* g = bsrc + (size_t)(kt + 1) * 1024 + tid * 8;
            asm volatile("global_load_async_to_lds_b64 %0, %1, off"
                         :: "v"(ldsn), "v"(g) : "memory");
            asm volatile("s_wait_asynccnt 0x1" ::: "memory");   // current tile done
        } else {
            asm volatile("s_wait_asynccnt 0x0" ::: "memory");
        }
        __syncthreads();                                         // tile visible to all waves

        v16h av = a[(size_t)kt * 32];
        if (kt + 1 < KT)
            __builtin_prefetch(&a[(size_t)(kt + 1) * 32], 0, 3); // near-scope prefetch
        v16h bv = *(const v16h*)&shB[kt & 1][lane * 16];

        acc = __builtin_amdgcn_wmma_f32_16x16x32_f16(
            /*neg_a=*/false, av, /*neg_b=*/false, bv,
            /*c_mod=*/(short)0, acc, /*reuse_a=*/false, /*reuse_b=*/false);

        __syncthreads();                       // all reads done before buffer reuse
    }

    int n = nt * 16 + (lane & 15);
    int mbase = mt * 16 + ((lane & 16) ? 8 : 0);
    for (int r = 0; r < 8; ++r) {
        int m = mbase + r;
        float v = acc[r] + bias[m];
        Y[(size_t)m * Ntot + n] = v > 0.f ? v : 0.f;
    }
}

// ---------------- BN stats: per-channel mean/rsqrt(var) + per-(b,c) sums ----
__global__ void bn_stats(const float* __restrict__ y3, float* __restrict__ sbc,
                         float* __restrict__ meanArr, float* __restrict__ invArr) {
    int c = blockIdx.x;       // 0..255
    int b = threadIdx.x;      // 0..63
    const float* p = y3 + (size_t)c * N3 + b * 9;
    float s = 0.f, q = 0.f;
    for (int t = 0; t < 9; ++t) { float v = p[t]; s += v; q += v * v; }
    sbc[b * Cc + c] = s;
    __shared__ float sh_s[64], sh_q[64];
    sh_s[b] = s; sh_q[b] = q;
    __syncthreads();
    for (int off = 32; off > 0; off >>= 1) {
        if (b < off) { sh_s[b] += sh_s[b + off]; sh_q[b] += sh_q[b + off]; }
        __syncthreads();
    }
    if (b == 0) {
        float mean = sh_s[0] / 576.f;
        float var  = sh_q[0] / 576.f - mean * mean;
        meanArr[c] = mean;
        invArr[c]  = rsqrtf(var + 1e-5f);
    }
}

// ---------------- final: BN affine + spatial mean + linear -------------------
__global__ void final_reduce(const float* __restrict__ sbc, const float* __restrict__ meanArr,
                             const float* __restrict__ invArr, const float* __restrict__ gamma,
                             const float* __restrict__ beta, const float* __restrict__ wl,
                             const float* __restrict__ bl, float* __restrict__ out) {
    int b = blockIdx.x;       // 0..63
    int c = threadIdx.x;      // 0..255
    float ms   = sbc[b * Cc + c] * (1.f / 9.f);
    float xhat = (ms - meanArr[c]) * invArr[c];
    float term = (xhat * gamma[c] + beta[c]) * wl[c];
    __shared__ float sh[256];
    sh[c] = term;
    __syncthreads();
    for (int off = 128; off > 0; off >>= 1) {
        if (c < off) sh[c] += sh[c + off];
        __syncthreads();
    }
    if (c == 0) out[b] = sh[0] + bl[0];
}

// ---------------- host side --------------------------------------------------
extern "C" void kernel_launch(void* const* d_in, const int* in_sizes, int n_in,
                              void* d_out, int out_size, void* d_ws, size_t ws_size,
                              hipStream_t stream) {
    const float* x_r   = (const float*)d_in[0];
    const float* x_l   = (const float*)d_in[1];
    const float* w1    = (const float*)d_in[2];
    const float* b1    = (const float*)d_in[3];
    const float* w2    = (const float*)d_in[4];
    const float* b2    = (const float*)d_in[5];
    const float* w3    = (const float*)d_in[6];
    const float* b3    = (const float*)d_in[7];
    const float* gamma = (const float*)d_in[8];
    const float* beta  = (const float*)d_in[9];
    const float* wl    = (const float*)d_in[10];
    const float* bl    = (const float*)d_in[11];

    char* ws = (char*)d_ws;
    auto carve = [&](size_t bytes) -> char* {
        char* p = ws;
        ws += (bytes + 255) & ~size_t(255);
        return p;
    };
    _Float16* A1p = (_Float16*)carve((size_t)Cc * K1 * 2);     // 3.4 MB
    _Float16* B1p = (_Float16*)carve((size_t)N1 * K1 * 2);     // 41.7 MB
    float*    Y1  = (float*)   carve((size_t)Cc * N1 * 4);     // 3.2 MB
    _Float16* A2p = (_Float16*)carve((size_t)Cc * K23 * 2);    // 1.2 MB
    _Float16* B2p = (_Float16*)carve((size_t)N2 * K23 * 2);    // 7.4 MB
    float*    Y2  = (float*)   carve((size_t)Cc * N2 * 4);     // 1.6 MB
    _Float16* A3p = (_Float16*)carve((size_t)Cc * K23 * 2);    // 1.2 MB
    _Float16* B3p = (_Float16*)carve((size_t)N3 * K23 * 2);    // 2.7 MB
    float*    Y3  = (float*)   carve((size_t)Cc * N3 * 4);     // 0.6 MB
    float*    sbc = (float*)   carve((size_t)Bb * Cc * 4);
    float*    mA  = (float*)   carve((size_t)Cc * 4);
    float*    iA  = (float*)   carve((size_t)Cc * 4);

    // Stage 1: fused cin_diff + conv1(5x5, stride5) as one GEMM (K = 26*C)
    pack_a1<<<(16 * KT1 * 32 * 16) / 256, 256, 0, stream>>>(w1, A1p);
    pack_b1<<<(NT1 * KT1 * 32 * 16) / 256, 256, 0, stream>>>(x_r, x_l, B1p);
    wmma_gemm_relu<<<dim3(4, NT1), 128, 0, stream>>>(A1p, B1p, b1, Y1, KT1, N1);

    // Stage 2: conv2 3x3 VALID (7x7 -> 5x5)
    pack_a_3x3<<<(16 * KT23 * 32 * 16) / 256, 256, 0, stream>>>(w2, A2p);
    pack_b2<<<(NT2 * KT23 * 32 * 16) / 256, 256, 0, stream>>>(Y1, B2p);
    wmma_gemm_relu<<<dim3(4, NT2), 128, 0, stream>>>(A2p, B2p, b2, Y2, KT23, N2);

    // Stage 3: conv3 3x3 VALID (5x5 -> 3x3)
    pack_a_3x3<<<(16 * KT23 * 32 * 16) / 256, 256, 0, stream>>>(w3, A3p);
    pack_b3<<<(NT3 * KT23 * 32 * 16) / 256, 256, 0, stream>>>(Y2, B3p);
    wmma_gemm_relu<<<dim3(4, NT3), 128, 0, stream>>>(A3p, B3p, b3, Y3, KT23, N3);

    // Stage 4: BN stats + fused BN/pool/linear
    bn_stats<<<Cc, 64, 0, stream>>>(Y3, sbc, mA, iA);
    final_reduce<<<Bb, 256, 0, stream>>>(sbc, mA, iA, gamma, beta, wl, bl, (float*)d_out);
}
